// MatrixKANLayer_43164421325572
// MI455X (gfx1250) — matrix operand order repaired and checked by reading, not compiled, not run on hardware
//
#include <hip/hip_runtime.h>
#include <hip/hip_bf16.h>
#include <stdint.h>

#define B_DIM   4096
#define IN_DIM  512
#define OUT_DIM 512
#define NUM_G   8
#define KORD    3
#define NB      11              // NUM + K spline basis columns
#define JW      12              // 11 spline cols + 1 silu/base col
#define KDIM    (IN_DIM * JW)   // 6144 GEMM K dimension

typedef __bf16 bf16;
typedef __attribute__((ext_vector_type(16))) __bf16 v16bf;
typedef __attribute__((ext_vector_type(8)))  float  v8f;
typedef __attribute__((ext_vector_type(4)))  unsigned int v4u;
typedef __attribute__((ext_vector_type(2)))  unsigned int v2u;

union FragU { v4u u[2]; v16bf v; };
union PairU { bf16 h[2]; unsigned int u; };

__device__ inline bf16 f2bf(float f) { return (bf16)f; }

// ---------------------------------------------------------------------------
// Kernel 1: expand x -> A[b, i*12 + j]  (bf16, row-major 4096 x 6144)
//   j in [0,10]: scattered cubic B-spline basis values; j==11: silu(x)
// Grid is uniform per the reference setup, so the interval is closed-form.
// ---------------------------------------------------------------------------
__global__ __launch_bounds__(256) void kan_expand(
    const float* __restrict__ X, const float* __restrict__ grid,
    const float* __restrict__ bm, bf16* __restrict__ A)
{
    int tid = blockIdx.x * 256 + threadIdx.x;
    if (tid >= B_DIM * IN_DIM) return;
    int i = tid % IN_DIM;

    float x  = X[tid];
    const float* g = grid + i * (NUM_G + 2 * KORD + 1);
    float g0 = g[0];
    float h  = (g[NUM_G + KORD] - g[KORD]) * (1.0f / NUM_G);
    float rel = (x - g0) / h;
    int   t   = (int)floorf(rel);
    bool  hit = (t >= 0) && (t < NUM_G + 2 * KORD);
    int   tc  = hit ? t : 0;

    float u1 = (x - (tc * h + g0)) / h;
    float u2 = u1 * u1;
    float u3 = u2 * u1;

    float out[JW];
#pragma unroll
    for (int j = 0; j < JW; j++) out[j] = 0.0f;

#pragma unroll
    for (int q = 0; q < 4; q++) {
        float lq = bm[0 * 4 + q] + u1 * bm[1 * 4 + q]
                 + u2 * bm[2 * 4 + q] + u3 * bm[3 * 4 + q];
        int idx = tc - KORD + q;
        if (hit && idx >= 0 && idx < NB) out[idx] = lq;
    }
    out[NB] = x / (1.0f + expf(-x));   // silu base activation

    unsigned int pk[6];
#pragma unroll
    for (int j = 0; j < 6; j++) {
        PairU c; c.h[0] = f2bf(out[2 * j]); c.h[1] = f2bf(out[2 * j + 1]);
        pk[j] = c.u;
    }
    v2u* dst = (v2u*)(A + (size_t)tid * JW);   // 24-byte stride, 8B aligned
    v2u t0; t0.x = pk[0]; t0.y = pk[1]; dst[0] = t0;
    v2u t1; t1.x = pk[2]; t1.y = pk[3]; dst[1] = t1;
    v2u t2; t2.x = pk[4]; t2.y = pk[5]; dst[2] = t2;
}

// ---------------------------------------------------------------------------
// Kernel 2: fold scales into Wt[o, i*12 + j] (bf16, row-major 512 x 6144)
//   so that B-matrix fragments are contiguous along K for each output column.
// ---------------------------------------------------------------------------
__global__ __launch_bounds__(256) void kan_weights(
    const float* __restrict__ coef, const float* __restrict__ scale_base,
    const float* __restrict__ scale_sp, const float* __restrict__ mask,
    bf16* __restrict__ Wt)
{
    int tid = blockIdx.x * 256 + threadIdx.x;
    if (tid >= OUT_DIM * IN_DIM) return;
    int o = tid / IN_DIM;
    int i = tid % IN_DIM;

    int   io  = i * OUT_DIM + o;
    float m   = mask[io];
    float ssp = scale_sp[io] * m;
    float sb  = scale_base[io] * m;
    const float* cp = coef + (size_t)io * NB;

    float out[JW];
#pragma unroll
    for (int j = 0; j < NB; j++) out[j] = cp[j] * ssp;
    out[NB] = sb;

    unsigned int pk[6];
#pragma unroll
    for (int j = 0; j < 6; j++) {
        PairU c; c.h[0] = f2bf(out[2 * j]); c.h[1] = f2bf(out[2 * j + 1]);
        pk[j] = c.u;
    }
    v2u* dst = (v2u*)(Wt + (size_t)o * KDIM + (size_t)i * JW);
    v2u t0; t0.x = pk[0]; t0.y = pk[1]; dst[0] = t0;
    v2u t1; t1.x = pk[2]; t1.y = pk[3]; dst[1] = t1;
    v2u t2; t2.x = pk[4]; t2.y = pk[5]; dst[2] = t2;
}

// ---------------------------------------------------------------------------
// Kernel 3: Y = A @ Wt^T  via v_wmma_f32_16x16x32_bf16 (wave32)
//   Wave tile 64(M) x 32(N) = 4x2 fragments; block = 8 waves -> 128 x 128.
//   A fragment (16x32 bf16): lane L holds row M=L&15; K-halves 0/8 by L>=16,
//   8 contiguous K in v0..3 and K+16..K+23 in v4..7  -> two b128 loads.
//   B fragment (32x16 bf16): lane L holds col N=L&15; K-halves 0/8 by L>=16,
//   8 contiguous K in v0..3 and K+16..K+23 in v4..7  -> two b128 loads.
// ---------------------------------------------------------------------------
__global__ __launch_bounds__(256) void kan_gemm(
    const bf16* __restrict__ A, const bf16* __restrict__ Wt,
    float* __restrict__ Y)
{
    const int lane  = threadIdx.x & 31;
    const int wave  = threadIdx.x >> 5;
    const int waveM = wave & 1;            // 2 waves along M
    const int waveN = wave >> 1;           // 4 waves along N
    const int m0 = blockIdx.x * 128 + waveM * 64;
    const int n0 = blockIdx.y * 128 + waveN * 32;

    v8f acc[4][2];
#pragma unroll
    for (int mi = 0; mi < 4; mi++)
#pragma unroll
        for (int ni = 0; ni < 2; ni++)
#pragma unroll
            for (int v = 0; v < 8; v++) acc[mi][ni][v] = 0.0f;

    const int l15 = lane & 15;
    const int ak  = (lane >> 4) << 3;      // A K-offset: 0 or 8
    const int bk  = (lane >> 4) << 3;      // B K-offset: 0 or 8

    const bf16* aptr = A  + (size_t)(m0 + l15) * KDIM + ak;
    const bf16* bptr = Wt + (size_t)(n0 + l15) * KDIM + bk;

    for (int k0 = 0; k0 < KDIM; k0 += 32) {
        FragU a[4], b[2];
#pragma unroll
        for (int mi = 0; mi < 4; mi++) {
            const bf16* p = aptr + (size_t)mi * 16 * KDIM + k0;
            a[mi].u[0] = *(const v4u*)p;          // K: ak .. ak+7
            a[mi].u[1] = *(const v4u*)(p + 16);   // K: ak+16 .. ak+23
        }
#pragma unroll
        for (int ni = 0; ni < 2; ni++) {
            const bf16* p = bptr + (size_t)ni * 16 * KDIM + k0;
            b[ni].u[0] = *(const v4u*)p;          // K: bk .. bk+7
            b[ni].u[1] = *(const v4u*)(p + 16);   // K: bk+16 .. bk+23
        }
#pragma unroll
        for (int mi = 0; mi < 4; mi++)
#pragma unroll
            for (int ni = 0; ni < 2; ni++)
                acc[mi][ni] = __builtin_amdgcn_wmma_f32_16x16x32_bf16(
                    false, a[mi].v, false, b[ni].v,
                    (short)0, acc[mi][ni], false, false);
    }

    // C/D layout: VGPR v, lanes 0-15 -> (M=v, N=lane); lanes 16-31 -> (M=v+8)
    const int mb = (lane >> 4) << 3;
#pragma unroll
    for (int mi = 0; mi < 4; mi++)
#pragma unroll
        for (int ni = 0; ni < 2; ni++) {
            int ncol = n0 + ni * 16 + l15;
#pragma unroll
            for (int v = 0; v < 8; v++) {
                int mrow = m0 + mi * 16 + mb + v;
                Y[(size_t)mrow * OUT_DIM + ncol] = acc[mi][ni][v];
            }
        }
}

extern "C" void kernel_launch(void* const* d_in, const int* in_sizes, int n_in,
                              void* d_out, int out_size, void* d_ws, size_t ws_size,
                              hipStream_t stream) {
    const float* X          = (const float*)d_in[0];
    const float* grid       = (const float*)d_in[1];
    const float* coef       = (const float*)d_in[2];
    const float* scale_base = (const float*)d_in[3];
    const float* scale_sp   = (const float*)d_in[4];
    const float* mask       = (const float*)d_in[5];
    const float* bm         = (const float*)d_in[6];
    float* Y = (float*)d_out;

    bf16* A  = (bf16*)d_ws;                                       // 48 MB
    bf16* Wt = (bf16*)((char*)d_ws + (size_t)B_DIM * KDIM * 2);   // +6 MB

    kan_expand<<<(B_DIM * IN_DIM + 255) / 256, 256, 0, stream>>>(X, grid, bm, A);
    kan_weights<<<(OUT_DIM * IN_DIM + 255) / 256, 256, 0, stream>>>(
        coef, scale_base, scale_sp, mask, Wt);

    dim3 g(B_DIM / 128, OUT_DIM / 128);
    kan_gemm<<<g, 256, 0, stream>>>(A, Wt, Y);
}
